// CropConvLSTM_51496657879766
// MI455X (gfx1250) — matrix operand order repaired and checked by reading, not compiled, not run on hardware
//
#include <hip/hip_runtime.h>
#include <stdint.h>

// ---------------- types for WMMA builtins (probe-confirmed shapes) ----------
typedef __attribute__((ext_vector_type(16))) __bf16 v16bf;
typedef __attribute__((ext_vector_type(8)))  float  v8f;
typedef __attribute__((ext_vector_type(2)))  float  v2f;
typedef int v4i_g __attribute__((vector_size(16)));   // gcc-style int4 (matches builtin)

#define B_  256
#define S_  12
#define L_  128
#define H_  64
#define C_  10
#define HS_ (H_ * S_)          // 768
#define K1_ (H_ * S_ * L_)     // 98304
#define EPS 1e-5f

#define KSPLIT 4
#define KSEG   (K1_ / KSPLIT)  // 24576
#define KC     128             // K-chunk staged per iteration
#define NCH    (KSEG / KC)     // 192
#define LDW    132             // padded LDS row stride (floats): conflict-free, 16B-aligned

#if __has_builtin(__builtin_amdgcn_global_load_async_to_lds_b128)
#define USE_ASYNC 1
#else
#define USE_ASYNC 0
#endif

__device__ __forceinline__ unsigned short f2bf(float f) {
  uint32_t u = __float_as_uint(f);
  uint32_t r = u + 0x7FFFu + ((u >> 16) & 1u);   // round-to-nearest-even
  return (unsigned short)(r >> 16);
}
__device__ __forceinline__ float sigf(float x) { return 1.f / (1.f + __expf(-x)); }

__device__ __forceinline__ void async_cp16(const float* g, float* l) {
#if USE_ASYNC
  __builtin_amdgcn_global_load_async_to_lds_b128(
      (__attribute__((address_space(1))) v4i_g*)g,
      (__attribute__((address_space(3))) v4i_g*)l, 0, 0);
#else
  *(float4*)l = *(const float4*)g;
#endif
}
__device__ __forceinline__ void wait_async_le(int n) {
#if USE_ASYNC
  asm volatile("s_wait_asynccnt %0" :: "i"(n) : "memory");
#else
  (void)n;
#endif
}

// ============================================================================
// Kernel A: ConvLSTM recurrence. One workgroup per batch element b.
// State: hT (bf16, transposed, halo-padded) in LDS; c in registers.
// cc = conv1d([x;h], W) done as 3 (per k-offset) bf16 WMMA GEMMs over K=64.
// ============================================================================
__launch_bounds__(256)
__global__ void convlstm_kernel(const float* __restrict__ x,
                                const float* __restrict__ conv_w,   // (256,65,3)
                                const float* __restrict__ conv_b,   // (256,)
                                float* __restrict__ hs)             // (B,S,H,L)
{
  __shared__ __align__(16) unsigned short lWb[3][256][64]; // bf16 W[k][oc][hc], 96 KB
  __shared__ __align__(16) unsigned short lHt[130][64];    // bf16 h^T, rows = l+1 (halo)
  __shared__ float  lWx[256][3];                           // x-channel weights
  __shared__ float  lBias[256];
  __shared__ float  lX[132];                               // x_t with halo
  __shared__ float  lHout[H_ * L_];                        // h_t as [hu][l] f32, 32 KB

  const int tid  = threadIdx.x;
  const int b    = blockIdx.x;
  const int lane = tid & 31;
  const int wv   = tid >> 5;        // 0..7 == l-tile index
  const int half = lane >> 4;       // 0/1
  const int ln   = lane & 15;

  // ---- stage weights (one time per workgroup) ----
  for (int i = tid; i < 3 * 256 * 64; i += 256) {
    int koff = i >> 14;
    int rem  = i & 16383;
    int oc   = rem >> 6;
    int hc   = rem & 63;
    lWb[koff][oc][hc] = f2bf(conv_w[oc * 195 + (1 + hc) * 3 + koff]);
  }
  for (int i = tid; i < 256 * 3; i += 256) {
    int oc = i / 3, k = i % 3;
    lWx[oc][k] = conv_w[oc * 195 + k];
  }
  if (tid < 256) lBias[tid] = conv_b[tid];
  for (int i = tid; i < 130 * 64; i += 256) ((unsigned short*)lHt)[i] = 0;

  float cReg[4][8];
#pragma unroll
  for (int j = 0; j < 4; ++j)
#pragma unroll
    for (int r = 0; r < 8; ++r) cReg[j][r] = 0.f;

  __syncthreads();

  for (int t = 0; t < S_; ++t) {
    __syncthreads();   // (a) prior h writes / lHout reads done
    if (tid < L_) lX[tid + 1] = x[(b * S_ + t) * L_ + tid];
    if (tid == 0) { lX[0] = 0.f; lX[L_ + 1] = 0.f; }

    // ---- load A operands: h patches, 16x32 bf16 per (koff,kchunk) ----
    v16bf a[3][2];
#pragma unroll
    for (int koff = 0; koff < 3; ++koff)
#pragma unroll
      for (int kc = 0; kc < 2; ++kc) {
        int row = wv * 16 + ln + koff;             // padded index l + koff
        int kb  = kc * 32 + (half ? 8 : 0);        // A: K0..K0+7 then K0+16..K0+23
        union { uint4 u[2]; v16bf v; } up;
        up.u[0] = *(const uint4*)&lHt[row][kb];
        up.u[1] = *(const uint4*)&lHt[row][kb + 16];
        a[koff][kc] = up.v;
      }
    __syncthreads();   // (b) lX visible; A reads done before h writes below

#pragma unroll 1
    for (int j = 0; j < 4; ++j) {                  // hidden-unit group (16 hu)
      v8f acc[4];
#pragma unroll
      for (int g = 0; g < 4; ++g) acc[g] = (v8f){0.f,0.f,0.f,0.f,0.f,0.f,0.f,0.f};

#pragma unroll
      for (int g = 0; g < 4; ++g) {                // gate: i,f,o,g
        int ocb = g * 64 + j * 16;
#pragma unroll
        for (int koff = 0; koff < 3; ++koff)
#pragma unroll
          for (int kc = 0; kc < 2; ++kc) {
            int kb = kc * 32 + (half ? 8 : 0);     // B: K0..K0+7 then K0+16..K0+23
            union { uint4 u[2]; v16bf v; } bp;
            bp.u[0] = *(const uint4*)&lWb[koff][ocb + ln][kb];
            bp.u[1] = *(const uint4*)&lWb[koff][ocb + ln][kb + 16];
            acc[g] = __builtin_amdgcn_wmma_f32_16x16x32_bf16(
                false, a[koff][kc], false, bp.v, (short)0, acc[g], false, false);
          }
      }

      // ---- gate math: D element (M = r + 8*half, N = ln) ----
#pragma unroll
      for (int r = 0; r < 8; ++r) {
        int l  = wv * 16 + r + half * 8;
        int hu = j * 16 + ln;
        float x0 = lX[l], x1 = lX[l + 1], x2 = lX[l + 2];
        int oci = hu, ocf = 64 + hu, oco = 128 + hu, ocg = 192 + hu;
        float vi = acc[0][r] + lBias[oci] + lWx[oci][0]*x0 + lWx[oci][1]*x1 + lWx[oci][2]*x2;
        float vf = acc[1][r] + lBias[ocf] + lWx[ocf][0]*x0 + lWx[ocf][1]*x1 + lWx[ocf][2]*x2;
        float vo = acc[2][r] + lBias[oco] + lWx[oco][0]*x0 + lWx[oco][1]*x1 + lWx[oco][2]*x2;
        float vg = acc[3][r] + lBias[ocg] + lWx[ocg][0]*x0 + lWx[ocg][1]*x1 + lWx[ocg][2]*x2;
        float ig = sigf(vi), fg = sigf(vf), og = sigf(vo), gg = tanhf(vg);
        float cn = fg * cReg[j][r] + ig * gg;
        cReg[j][r] = cn;
        float hn = og * tanhf(cn);
        lHt[l + 1][hu]     = f2bf(hn);             // next-step WMMA operand
        lHout[hu * L_ + l] = hn;                   // transposed for coalesced store
      }
    }
    __syncthreads();   // (c) lHout complete

    float* dst = hs + ((size_t)(b * S_ + t) << 13);   // (b,s,hu,l), 8192 f32
    for (int i = tid; i < H_ * L_; i += 256) dst[i] = lHout[i];
  }
}

// ============================================================================
// Kernel B: partial GEMM1 (flat @ w1^T), f32 WMMA 16x16x4.
// grid = (16 b-tiles, 6 of-groups of 128, KSPLIT), 256 threads (8 waves).
// Double-buffered LDS tiles filled by async global->LDS DMA; padded rows
// (LDW=132) make every WMMA operand ds_load_b64 bank-conflict-free.
// ============================================================================
__launch_bounds__(256)
__global__ void gemm1_kernel(const float* __restrict__ flat,  // (256, 98304)
                             const float* __restrict__ w1,    // (768, 98304)
                             float* __restrict__ part)        // (KSPLIT, 256, 768)
{
  __shared__ __align__(16) float lA[2][16 * LDW];    // 2 x 8.25 KB
  __shared__ __align__(16) float lW[2][128 * LDW];   // 2 x 66 KB

  const int tid  = threadIdx.x;
  const int lane = tid & 31;
  const int wv   = tid >> 5;        // 0..7: of-tile within 128-col group
  const int half = lane >> 4;
  const int ln   = lane & 15;
  const int bt   = blockIdx.x;      // 0..15
  const int og   = blockIdx.y;      // 0..5
  const int kz   = blockIdx.z;      // 0..KSPLIT-1
  const int kbase = kz * KSEG;

  const float* gA = flat + (size_t)(bt * 16) * K1_ + kbase;
  const float* gW = w1 + (size_t)(og * 128) * K1_ + kbase;

  // per-thread staging (float4 granularity):
  // lA: 16 rows x 32 float4 = 512 -> 2 per thread; lW: 128 x 32 = 4096 -> 16
  auto issue_chunk = [&](int buf, int k0) {
#pragma unroll
    for (int i = 0; i < 2; ++i) {
      int idx4 = tid + i * 256;
      int r = idx4 >> 5, c4 = idx4 & 31;
      async_cp16(gA + (size_t)r * K1_ + k0 + c4 * 4, &lA[buf][r * LDW + c4 * 4]);
    }
#pragma unroll
    for (int i = 0; i < 16; ++i) {
      int idx4 = tid + i * 256;
      int r = idx4 >> 5, c4 = idx4 & 31;
      async_cp16(gW + (size_t)r * K1_ + k0 + c4 * 4, &lW[buf][r * LDW + c4 * 4]);
    }
  };

  v8f acc = (v8f){0.f,0.f,0.f,0.f,0.f,0.f,0.f,0.f};

  issue_chunk(0, 0);
  for (int c = 0; c < NCH; ++c) {
    if (c + 1 < NCH) issue_chunk((c + 1) & 1, (c + 1) * KC);
    wait_async_le((c + 1 < NCH) ? 18 : 0);   // 18 = ops of the in-flight chunk
    __syncthreads();                          // all waves' portions visible
    const float* cA = lA[c & 1];
    const float* cW = lW[c & 1];
#pragma unroll
    for (int ks = 0; ks < KC; ks += 4) {
      // A 16x4 f32: lane row = ln, K pair at half*2. B 4x16: lane col = ln.
      v2f av = *(const v2f*)&cA[ln * LDW + ks + half * 2];
      v2f bv = *(const v2f*)&cW[(wv * 16 + ln) * LDW + ks + half * 2];
      acc = __builtin_amdgcn_wmma_f32_16x16x4_f32(false, av, false, bv,
                                                  (short)0, acc, false, false);
    }
    __syncthreads();   // done reading before DMA overwrites this buffer
  }

  int of = og * 128 + wv * 16 + ln;
  float* P = part + ((size_t)kz * B_) * HS_;
#pragma unroll
  for (int r = 0; r < 8; ++r) {
    int brow = bt * 16 + r + half * 8;
    P[(size_t)brow * HS_ + of] = acc[r];
  }
}

// ============================================================================
// Kernel B2: reduce K-split partials + bias + BN + ReLU -> y1 (256,768)
// ============================================================================
__launch_bounds__(256)
__global__ void reduce_bn_kernel(const float* __restrict__ part,
                                 const float* __restrict__ b1, const float* __restrict__ g1,
                                 const float* __restrict__ be1, const float* __restrict__ m1,
                                 const float* __restrict__ v1,
                                 float* __restrict__ y1)
{
  int idx = blockIdx.x * 256 + threadIdx.x;     // 0 .. 256*768-1
  int of = idx % HS_;
  float s = 0.f;
#pragma unroll
  for (int z = 0; z < KSPLIT; ++z) s += part[(size_t)z * B_ * HS_ + idx];
  float y = (s + b1[of] - m1[of]) * rsqrtf(v1[of] + EPS) * g1[of] + be1[of];
  y1[idx] = fmaxf(y, 0.f);
}

// ============================================================================
// Kernel C: y2 = relu(BN(y1 @ w2^T + b2)); out = y2 @ w3^T + b3.  (tiny)
// ============================================================================
__launch_bounds__(64)
__global__ void head_tail_kernel(const float* __restrict__ y1,
                                 const float* __restrict__ w2, const float* __restrict__ b2,
                                 const float* __restrict__ g2, const float* __restrict__ be2,
                                 const float* __restrict__ m2, const float* __restrict__ v2,
                                 const float* __restrict__ w3, const float* __restrict__ b3,
                                 float* __restrict__ out)
{
  __shared__ float lw2[S_ * HS_];   // 36 KB
  const int tid = threadIdx.x;
  for (int i = tid; i < S_ * HS_; i += 64) lw2[i] = w2[i];
  __syncthreads();

  int b = blockIdx.x * 64 + tid;    // 0..255
  const float* yr = y1 + (size_t)b * HS_;
  float y2[S_];
#pragma unroll 1
  for (int s = 0; s < S_; ++s) {
    float acc = 0.f;
    const float* wr = lw2 + s * HS_;
    for (int k = 0; k < HS_; ++k) acc += yr[k] * wr[k];
    acc = (acc + b2[s] - m2[s]) * rsqrtf(v2[s] + EPS) * g2[s] + be2[s];
    y2[s] = fmaxf(acc, 0.f);
  }
#pragma unroll
  for (int c = 0; c < C_; ++c) {
    float acc = b3[c];
#pragma unroll
    for (int s = 0; s < S_; ++s) acc += y2[s] * w3[c * S_ + s];
    out[b * C_ + c] = acc;
  }
}

// ============================================================================
extern "C" void kernel_launch(void* const* d_in, const int* in_sizes, int n_in,
                              void* d_out, int out_size, void* d_ws, size_t ws_size,
                              hipStream_t stream) {
  const float* x      = (const float*)d_in[0];
  const float* conv_w = (const float*)d_in[1];
  const float* conv_b = (const float*)d_in[2];
  const float* w1  = (const float*)d_in[3];
  const float* b1  = (const float*)d_in[4];
  const float* g1  = (const float*)d_in[5];
  const float* be1 = (const float*)d_in[6];
  const float* m1  = (const float*)d_in[7];
  const float* v1  = (const float*)d_in[8];
  const float* w2  = (const float*)d_in[9];
  const float* b2  = (const float*)d_in[10];
  const float* g2  = (const float*)d_in[11];
  const float* be2 = (const float*)d_in[12];
  const float* m2  = (const float*)d_in[13];
  const float* v2  = (const float*)d_in[14];
  const float* w3  = (const float*)d_in[15];
  const float* b3  = (const float*)d_in[16];

  float* hs   = (float*)d_ws;                          // 256*98304 f32 (~100.7 MB)
  float* part = hs + (size_t)B_ * K1_;                 // KSPLIT*256*768 f32 (3 MB)
  float* y1   = part + (size_t)KSPLIT * B_ * HS_;      // 256*768 f32

  convlstm_kernel<<<B_, 256, 0, stream>>>(x, conv_w, conv_b, hs);
  gemm1_kernel<<<dim3(16, 6, KSPLIT), 256, 0, stream>>>(hs, w1, part);
  reduce_bn_kernel<<<(B_ * HS_) / 256, 256, 0, stream>>>(part, b1, g1, be1, m1, v1, y1);
  head_tail_kernel<<<4, 64, 0, stream>>>(y1, w2, b2, g2, be2, m2, v2, w3, b3,
                                         (float*)d_out);
}
